// Encoder_7541962572066
// MI455X (gfx1250) — compile-verified, matrix operand order checked by reading, not run on hardware
//
#include <hip/hip_runtime.h>
#include <math.h>

typedef __attribute__((ext_vector_type(4)))  float  v4f;
typedef __attribute__((ext_vector_type(8)))  float  v8f;
typedef __attribute__((ext_vector_type(8)))  __bf16 v8bf;
typedef __attribute__((ext_vector_type(16))) __bf16 v16bf;

#define B_  32
#define L_  256
#define D_  512
#define H_  8
#define DK_ 64
#define DV_ 64
#define NL_ 6
#define DI_ 2048
#define NA_ 64

// ---------------------------------------------------------------------------
// Generic batched GEMM: C = epi(alpha * A(f32->bf16) x B(f32->bf16) + bias) + resid
// A: M x K (row major, lda), B: K x N (row major, ldb) or N x K if transB.
// Requirements used by this workload: M % 128 == 0, K % 32 == 0, N % 8 == 0.
// LDS tiles are stored pre-swizzled in WMMA fragment order so each lane's
// 16-element bf16 fragment is 32 contiguous bytes (ds_load_b128 x2).
// ---------------------------------------------------------------------------
#define BM 128
#define BN 64
#define BK 32

__launch_bounds__(256)
__global__ void gemm_bf16_wmma(
    const float* __restrict__ A, const float* __restrict__ Bm, float* __restrict__ C,
    const float* __restrict__ bias, const float* __restrict__ resid,
    int M, int N, int K, int lda, int ldb, int ldc,
    long long sAb, long long sAh, long long sBb, long long sBh,
    long long sCb, long long sCh,
    int batchH, int transB, int epi, float alpha)
{
    int z  = blockIdx.z;
    int bb = z / batchH;
    int bh = z - bb * batchH;
    const float* Ab = A  + bb * sAb + bh * sAh;
    const float* Bb = Bm + bb * sBb + bh * sBh;
    float*       Cb = C  + bb * sCb + bh * sCh;
    const float* Rb = resid ? (resid + bb * sCb + bh * sCh) : (const float*)0;

    // Fragment-order LDS: [subtile][lane][elem_block(8 bf16)]
    __shared__ v8bf Asf[8][32][2];   // 8 KB : 8 m-subtiles of 16x32
    __shared__ v8bf Bsf[4][32][2];   // 4 KB : 4 n-subtiles of 32x16

    int m0   = blockIdx.y * BM;
    int n0   = blockIdx.x * BN;
    int tid  = threadIdx.x;
    int wave = tid >> 5;             // 8 waves (wave32)
    int lane = tid & 31;
    int lm   = lane & 15;
    int lh   = lane >> 4;
    int wm   = m0 + (wave << 4);     // this wave's 16-row strip

    // ---- staging maps (all uniform per thread) ----
    // A: 2 chunks/thread; chunk = tid + i*256 ; r = chunk>>2 (0..127), cb = chunk&3
    //    chunk covers row m0+r, K cols k0+cb*8 .. +7  -> Asf[r>>4][(r&15)+16*(cb&1)][cb>>1]
    const int ar[2] = { (tid + 0) >> 2, (tid + 256) >> 2 };
    const int acb[2] = { tid & 3, tid & 3 };
    // B transB: n = tid>>2 (0..63), kb = tid&3   (contiguous K in memory)
    // B normal: r = tid>>3 (0..31), nb = tid&7   (contiguous N in memory)
    const int btn = tid >> 2, btk = tid & 3;
    const int bnr = tid >> 3, bnb = tid & 7;

    v4f pa[2][2];   // prefetched A (2 chunks x 8 floats)
    v4f pb[2];      // prefetched B (1 chunk x 8 floats)

    // ---- prologue: load tile k0 = 0 into registers ----
    {
        #pragma unroll
        for (int i = 0; i < 2; ++i) {
            const v4f* p = (const v4f*)(Ab + (long long)(m0 + ar[i]) * lda + acb[i] * 8);
            pa[i][0] = p[0]; pa[i][1] = p[1];
        }
        if (transB) {
            if (n0 + btn < N) {
                const v4f* p = (const v4f*)(Bb + (long long)(n0 + btn) * ldb + btk * 8);
                pb[0] = p[0]; pb[1] = p[1];
            } else { pb[0] = (v4f)0.f; pb[1] = (v4f)0.f; }
        } else {
            if (n0 + bnb * 8 < N) {
                const v4f* p = (const v4f*)(Bb + (long long)bnr * ldb + n0 + bnb * 8);
                pb[0] = p[0]; pb[1] = p[1];
            } else { pb[0] = (v4f)0.f; pb[1] = (v4f)0.f; }
        }
    }

    v8f acc[4] = {};

    for (int k0 = 0; k0 < K; k0 += BK) {
        // ---- commit prefetched registers to LDS (convert f32 -> bf16) ----
        #pragma unroll
        for (int i = 0; i < 2; ++i) {
            v8bf h;
            #pragma unroll
            for (int j = 0; j < 4; ++j) {
                h[j]     = (__bf16)pa[i][0][j];
                h[j + 4] = (__bf16)pa[i][1][j];
            }
            Asf[ar[i] >> 4][(ar[i] & 15) + ((acb[i] & 1) << 4)][acb[i] >> 1] = h;
        }
        if (transB) {
            v8bf h;
            #pragma unroll
            for (int j = 0; j < 4; ++j) {
                h[j]     = (__bf16)pb[0][j];
                h[j + 4] = (__bf16)pb[1][j];
            }
            Bsf[btn >> 4][(btn & 15) + ((btk & 1) << 4)][btk >> 1] = h;
        } else {
            __bf16* bs = (__bf16*)Bsf;
            int lh2  = (bnr >> 3) & 1;
            int elem = (bnr & 7) + ((bnr >> 4) << 3);
            #pragma unroll
            for (int j = 0; j < 8; ++j) {
                int n = bnb * 8 + j;
                float v = (j < 4) ? pb[0][j] : pb[1][j - 4];
                bs[((n >> 4) * 32 + (n & 15) + (lh2 << 4)) * 16 + elem] = (__bf16)v;
            }
        }
        __syncthreads();

        // ---- prefetch next tile (overlaps with WMMA below) ----
        int kn = k0 + BK;
        if (kn < K) {
            #pragma unroll
            for (int i = 0; i < 2; ++i) {
                const v4f* p = (const v4f*)(Ab + (long long)(m0 + ar[i]) * lda + kn + acb[i] * 8);
                pa[i][0] = p[0]; pa[i][1] = p[1];
            }
            if (transB) {
                if (n0 + btn < N) {
                    const v4f* p = (const v4f*)(Bb + (long long)(n0 + btn) * ldb + kn + btk * 8);
                    pb[0] = p[0]; pb[1] = p[1];
                } else { pb[0] = (v4f)0.f; pb[1] = (v4f)0.f; }
            } else {
                if (n0 + bnb * 8 < N) {
                    const v4f* p = (const v4f*)(Bb + (long long)(kn + bnr) * ldb + n0 + bnb * 8);
                    pb[0] = p[0]; pb[1] = p[1];
                } else { pb[0] = (v4f)0.f; pb[1] = (v4f)0.f; }
            }
        }

        // ---- WMMA: A fragment is 32 contiguous bytes per lane ----
        v8bf a0 = Asf[wave][lane][0];
        v8bf a1 = Asf[wave][lane][1];
        v16bf af = __builtin_shufflevector(a0, a1, 0,1,2,3,4,5,6,7,8,9,10,11,12,13,14,15);
        #pragma unroll
        for (int t = 0; t < 4; ++t) {
            v8bf b0 = Bsf[t][lane][0];
            v8bf b1 = Bsf[t][lane][1];
            v16bf bfm = __builtin_shufflevector(b0, b1, 0,1,2,3,4,5,6,7,8,9,10,11,12,13,14,15);
            acc[t] = __builtin_amdgcn_wmma_f32_16x16x32_bf16(
                false, af, false, bfm, (short)0, acc[t], false, false);
        }
        __syncthreads();
    }

    // ---- epilogue. C/D layout: elem r -> row r (+8 for hi lanes), col = lane&15 ----
    #pragma unroll
    for (int t = 0; t < 4; ++t) {
        int col = n0 + (t << 4) + lm;
        if (col < N) {
            #pragma unroll
            for (int r = 0; r < 8; ++r) {
                int row = wm + r + (lh << 3);
                float v = alpha * acc[t][r];
                if (bias) v += bias[col];
                if (epi == 2) {           // tanh gelu (jax.nn.gelu default)
                    float x3 = v * v * v;
                    v = 0.5f * v * (1.f + tanhf(0.7978845608028654f * (v + 0.044715f * x3)));
                } else if (epi == 3) {    // relu
                    v = fmaxf(v, 0.f);
                }
                long long o = (long long)row * ldc + col;
                if (Rb) v += Rb[o];
                Cb[o] = v;
            }
        }
    }
}

// ---------------------------------------------------------------------------
// Gaussian edge features: edge[b,i,j,d], masked on j; one block per (b,i,j)
// ---------------------------------------------------------------------------
__global__ void edge_kernel(const float* __restrict__ src_pos,
                            const float* __restrict__ g_means,
                            const float* __restrict__ g_stds,
                            const float* __restrict__ g_mul,
                            const float* __restrict__ g_bias,
                            float* __restrict__ edge)
{
    int idx = blockIdx.x;                 // (b*NA + i)*NA + j
    int j = idx % NA_;
    int bi = idx / NA_;                   // b*NA + i
    int b = bi / NA_;
    const float* pi = src_pos + (long long)bi * 3;
    const float* pj = src_pos + ((long long)b * NA_ + j) * 3;
    float dx = pi[0] - pj[0], dy = pi[1] - pj[1], dz = pi[2] - pj[2];
    float dist = sqrtf(dx * dx + dy * dy + dz * dz);
    bool maskj = (pj[0] == 0.f && pj[1] == 0.f && pj[2] == 0.f);
    float mul = g_mul[0], gb = g_bias[0];
    for (int d = threadIdx.x; d < D_; d += 256) {
        float sd = fabsf(g_stds[d]) + 1e-5f;
        float t = (mul * dist + gb - g_means[d]) / sd;
        float e = __expf(-0.5f * t * t) / (2.5066282746310002f * sd);
        edge[(long long)idx * D_ + d] = maskj ? 0.f : e;
    }
}

// node_feat[b,i,d] = sum_j edge[b,i,j,d]; one block per (b,i)
__global__ void nodefeat_kernel(const float* __restrict__ edge, float* __restrict__ node)
{
    long long bi = blockIdx.x;
    for (int d = threadIdx.x; d < D_; d += 256) {
        float s = 0.f;
        for (int j = 0; j < NA_; ++j)
            s += edge[(bi * NA_ + j) * D_ + d];
        node[bi * D_ + d] = s;
    }
}

// x0[b,l,d] = rate1*(emb[seq[b,l],d] + pos_table[l,d]); one block per (b,l)
__global__ void embed_kernel(const int* __restrict__ seq,
                             const float* __restrict__ emb,
                             const float* __restrict__ pos_table,
                             const float* __restrict__ rate1,
                             float* __restrict__ x)
{
    long long row = blockIdx.x;           // b*L + l
    int l = (int)(row % L_);
    int tok = seq[row];
    float r1 = rate1[0];
    for (int d = threadIdx.x; d < D_; d += 256)
        x[row * D_ + d] = r1 * (emb[(long long)tok * D_ + d] + pos_table[(long long)l * D_ + d]);
}

// x[b, atoms[b,a], :] += rate2 * node[b,a,:]; one block per (b,a)
__global__ void scatter_node_kernel(const int* __restrict__ atoms,
                                    const float* __restrict__ node,
                                    const float* __restrict__ rate2,
                                    float* __restrict__ x)
{
    long long ba = blockIdx.x;            // b*NA + a
    int b = (int)(ba / NA_);
    int row = b * L_ + atoms[ba];
    float r2 = rate2[0];
    for (int d = threadIdx.x; d < D_; d += 256)
        x[(long long)row * D_ + d] += r2 * node[ba * D_ + d];
}

// In-place layernorm over last dim D=512; one block (256 thr) per row
__global__ void layernorm_kernel(float* __restrict__ x,
                                 const float* __restrict__ g,
                                 const float* __restrict__ b)
{
    long long row = blockIdx.x;
    float* xr = x + row * D_;
    int tid = threadIdx.x;
    float a0 = xr[tid], a1 = xr[tid + 256];
    __shared__ float red[256];
    red[tid] = a0 + a1;
    __syncthreads();
    for (int s = 128; s > 0; s >>= 1) { if (tid < s) red[tid] += red[tid + s]; __syncthreads(); }
    float mean = red[0] * (1.f / D_);
    __syncthreads();
    float d0 = a0 - mean, d1 = a1 - mean;
    red[tid] = d0 * d0 + d1 * d1;
    __syncthreads();
    for (int s = 128; s > 0; s >>= 1) { if (tid < s) red[tid] += red[tid + s]; __syncthreads(); }
    float rs = rsqrtf(red[0] * (1.f / D_) + 1e-6f);
    xr[tid]       = d0 * rs * g[tid]       + b[tid];
    xr[tid + 256] = d1 * rs * g[tid + 256] + b[tid + 256];
}

// bias[b,h,atoms[b,i],atoms[b,j]] = ab[b,i,j,h]; flat over B*NA*NA*H
__global__ void scatter_bias_kernel(const int* __restrict__ atoms,
                                    const float* __restrict__ ab,
                                    float* __restrict__ bias)
{
    long long idx = (long long)blockIdx.x * blockDim.x + threadIdx.x;
    if (idx >= (long long)B_ * NA_ * NA_ * H_) return;
    int h = (int)(idx % H_);
    long long t = idx / H_;
    int j = (int)(t % NA_);
    t /= NA_;
    int i = (int)(t % NA_);
    int b = (int)(t / NA_);
    int ri = atoms[b * NA_ + i];
    int rj = atoms[b * NA_ + j];
    bias[(((long long)b * H_ + h) * L_ + ri) * L_ + rj] = ab[idx];
}

// scores += bias, mask (-1e9), row softmax in-place. One block per (b,h,q), 256 thr = L
__global__ void softmax_kernel(float* __restrict__ scores,
                               const float* __restrict__ bias,
                               const unsigned char* __restrict__ mask)
{
    long long rowIdx = blockIdx.x;        // ((b*H + h)*L + q)
    long long bh = rowIdx / L_;
    int b = (int)(bh / H_);
    float* s = scores + rowIdx * L_;
    const float* bi = bias + rowIdx * L_;
    int col = threadIdx.x;
    float v = s[col] + bi[col];
    if (!mask[(long long)b * L_ + col]) v = -1e9f;
    __shared__ float red[256];
    red[col] = v;
    __syncthreads();
    for (int st = 128; st > 0; st >>= 1) { if (col < st) red[col] = fmaxf(red[col], red[col + st]); __syncthreads(); }
    float mx = red[0];
    __syncthreads();
    float e = __expf(v - mx);
    red[col] = e;
    __syncthreads();
    for (int st = 128; st > 0; st >>= 1) { if (col < st) red[col] += red[col + st]; __syncthreads(); }
    s[col] = e / red[0];
}

// final: d_out[0:nx) = x, d_out[nx:nx+nm) = mask as float
__global__ void out_copy_kernel(const float* __restrict__ x,
                                const unsigned char* __restrict__ mask,
                                float* __restrict__ out, long long nx, long long nm)
{
    long long i = (long long)blockIdx.x * blockDim.x + threadIdx.x;
    if (i < nx)            out[i] = x[i];
    else if (i < nx + nm)  out[i] = mask[i - nx] ? 1.f : 0.f;
}

// ---------------------------------------------------------------------------
extern "C" void kernel_launch(void* const* d_in, const int* in_sizes, int n_in,
                              void* d_out, int out_size, void* d_ws, size_t ws_size,
                              hipStream_t stream)
{
    const int*           src_seq  = (const int*)d_in[0];
    const unsigned char* src_mask = (const unsigned char*)d_in[1];
    const float*         src_pos  = (const float*)d_in[2];
    const int*           atoms    = (const int*)d_in[4];
    const float* emb      = (const float*)d_in[5];
    const float* postab   = (const float*)d_in[6];
    const float* rate1    = (const float*)d_in[7];
    const float* rate2    = (const float*)d_in[8];
    const float* g_means  = (const float*)d_in[9];
    const float* g_stds   = (const float*)d_in[10];
    const float* g_mul    = (const float*)d_in[11];
    const float* g_bias   = (const float*)d_in[12];
    const float* proj1_w  = (const float*)d_in[13];
    const float* proj1_b  = (const float*)d_in[14];
    const float* proj2_w  = (const float*)d_in[15];
    const float* proj2_b  = (const float*)d_in[16];
    const float* ln_g     = (const float*)d_in[17];
    const float* ln_b     = (const float*)d_in[18];
    const float* Wq       = (const float*)d_in[19];
    const float* Wk       = (const float*)d_in[20];
    const float* Wv       = (const float*)d_in[21];
    const float* Wo       = (const float*)d_in[22];
    const float* ln1_g    = (const float*)d_in[23];
    const float* ln1_b    = (const float*)d_in[24];
    const float* W1       = (const float*)d_in[25];
    const float* b1       = (const float*)d_in[26];
    const float* W2       = (const float*)d_in[27];
    const float* b2       = (const float*)d_in[28];
    const float* ln2_g    = (const float*)d_in[29];
    const float* ln2_b    = (const float*)d_in[30];

    // ---- workspace layout (with reuse of the two big 268MB regions) ----
    char* ws = (char*)d_ws;
    size_t off = 0;
    auto take = [&](size_t n) { void* p = ws + off; off += (n + 255) & ~(size_t)255; return p; };

    const size_t edgeBytes = (size_t)B_ * NA_ * NA_ * D_ * 4;     // 268 MB
    float* edge  = (float*)take(edgeBytes);
    float* hproj = (float*)take(edgeBytes);                       // 268 MB
    float* node  = (float*)take((size_t)B_ * NA_ * D_ * 4);
    float* ab    = (float*)take((size_t)B_ * NA_ * NA_ * H_ * 4);
    float* biasM = (float*)take((size_t)B_ * H_ * L_ * L_ * 4);   // 67 MB
    float* bufX  = (float*)take((size_t)B_ * L_ * D_ * 4);        // 16 MB
    float* bufT  = (float*)take((size_t)B_ * L_ * D_ * 4);        // 16 MB

    // overlays (regions dead by the time these are used)
    float* scores = edge;                                         // 67 MB (attn, per layer)
    float* h1     = (float*)((char*)edge + ((size_t)67108864));   // 67 MB (ffn hidden)
    float* q      = hproj;                                        // 16 MB
    float* k      = (float*)((char*)hproj + (size_t)16777216);
    float* v      = (float*)((char*)hproj + (size_t)33554432);
    float* o      = (float*)((char*)hproj + (size_t)50331648);

    auto gemm = [&](const float* A, const float* Bm, float* C,
                    const float* bias, const float* resid,
                    int M, int N, int K, int lda, int ldb, int ldc,
                    long long sAb, long long sAh, long long sBb, long long sBh,
                    long long sCb, long long sCh,
                    int batchB, int batchH, int transB, int epi, float alpha) {
        dim3 grid((N + BN - 1) / BN, (M + BM - 1) / BM, batchB * batchH);
        gemm_bf16_wmma<<<grid, 256, 0, stream>>>(A, Bm, C, bias, resid,
            M, N, K, lda, ldb, ldc, sAb, sAh, sBb, sBh, sCb, sCh,
            batchH, transB, epi, alpha);
    };

    // ---- edge features + node feat + input embedding ----
    edge_kernel<<<B_ * NA_ * NA_, 256, 0, stream>>>(src_pos, g_means, g_stds, g_mul, g_bias, edge);
    nodefeat_kernel<<<B_ * NA_, 256, 0, stream>>>(edge, node);
    embed_kernel<<<B_ * L_, 256, 0, stream>>>(src_seq, emb, postab, rate1, bufX);
    scatter_node_kernel<<<B_ * NA_, 256, 0, stream>>>(atoms, node, rate2, bufX);
    layernorm_kernel<<<B_ * L_, 256, 0, stream>>>(bufX, ln_g, ln_b);

    // ---- edge MLP: hproj = gelu(edge@proj1 + b1p); ab = hproj@proj2 + b2p ----
    const int ME = B_ * NA_ * NA_;                    // 131072
    gemm(edge, proj1_w, hproj, proj1_b, nullptr,
         ME, D_, D_, D_, D_, D_, 0, 0, 0, 0, 0, 0, 1, 1, 0, /*gelu*/2, 1.f);
    gemm(hproj, proj2_w, ab, proj2_b, nullptr,
         ME, H_, D_, D_, H_, H_, 0, 0, 0, 0, 0, 0, 1, 1, 0, /*bias only*/0, 1.f);

    // ---- attention bias matrix (B,H,L,L): zero then scatter ----
    (void)hipMemsetAsync(biasM, 0, (size_t)B_ * H_ * L_ * L_ * 4, stream);
    {
        long long tot = (long long)B_ * NA_ * NA_ * H_;
        scatter_bias_kernel<<<(unsigned)((tot + 255) / 256), 256, 0, stream>>>(atoms, ab, biasM);
    }

    // ---- transformer layers ----
    const int M = B_ * L_;                            // 8192
    const float scale = 0.125f;                       // 1/sqrt(DK)
    for (int l = 0; l < NL_; ++l) {
        const float* wq = Wq + (size_t)l * D_ * (H_ * DK_);
        const float* wk = Wk + (size_t)l * D_ * (H_ * DK_);
        const float* wv = Wv + (size_t)l * D_ * (H_ * DV_);
        const float* wo = Wo + (size_t)l * (H_ * DV_) * D_;

        // q/k/v = x @ W   (laid out as (B,L,H,Dh) contiguous)
        gemm(bufX, wq, q, nullptr, nullptr, M, 512, D_, D_, 512, 512,
             0, 0, 0, 0, 0, 0, 1, 1, 0, 0, 1.f);
        gemm(bufX, wk, k, nullptr, nullptr, M, 512, D_, D_, 512, 512,
             0, 0, 0, 0, 0, 0, 1, 1, 0, 0, 1.f);
        gemm(bufX, wv, v, nullptr, nullptr, M, 512, D_, D_, 512, 512,
             0, 0, 0, 0, 0, 0, 1, 1, 0, 0, 1.f);

        // scores[b,h] = scale * q[b,h] @ k[b,h]^T   (batched over B*H)
        gemm(q, k, scores, nullptr, nullptr,
             L_, L_, DK_, 512, 512, L_,
             (long long)L_ * 512, DK_,                 // A strides (b,h)
             (long long)L_ * 512, DK_,                 // B strides (b,h)
             (long long)H_ * L_ * L_, (long long)L_ * L_,
             B_, H_, /*transB*/1, 0, scale);

        softmax_kernel<<<B_ * H_ * L_, 256, 0, stream>>>(scores, biasM, src_mask);

        // o[b,h] = attn[b,h] @ v[b,h]   -> stored as (B,L,H,DV)
        gemm(scores, v, o, nullptr, nullptr,
             L_, DV_, L_, L_, 512, 512,
             (long long)H_ * L_ * L_, (long long)L_ * L_,
             (long long)L_ * 512, DV_,
             (long long)L_ * 512, DV_,
             B_, H_, 0, 0, 1.f);

        // x1 = LN(o @ Wo + x)
        gemm(o, wo, bufT, nullptr, bufX, M, D_, H_ * DV_, 512, D_, D_,
             0, 0, 0, 0, 0, 0, 1, 1, 0, 0, 1.f);
        layernorm_kernel<<<M, 256, 0, stream>>>(bufT, ln1_g + (size_t)l * D_, ln1_b + (size_t)l * D_);

        // FFN: x = LN(relu(x1@W1 + b1) @ W2 + b2 + x1)
        gemm(bufT, W1 + (size_t)l * D_ * DI_, h1, b1 + (size_t)l * DI_, nullptr,
             M, DI_, D_, D_, DI_, DI_, 0, 0, 0, 0, 0, 0, 1, 1, 0, /*relu*/3, 1.f);
        gemm(h1, W2 + (size_t)l * DI_ * D_, bufX, b2 + (size_t)l * D_, bufT,
             M, D_, DI_, DI_, D_, D_, 0, 0, 0, 0, 0, 0, 1, 1, 0, 0, 1.f);
        layernorm_kernel<<<M, 256, 0, stream>>>(bufX, ln2_g + (size_t)l * D_, ln2_b + (size_t)l * D_);
    }

    // ---- outputs: x then mask (as float) ----
    long long nx = (long long)B_ * L_ * D_;
    long long nm = (long long)B_ * L_;
    out_copy_kernel<<<(unsigned)((nx + nm + 255) / 256), 256, 0, stream>>>(
        bufX, src_mask, (float*)d_out, nx, nm);
}
